// LongDistanceAttention_23210003268192
// MI455X (gfx1250) — matrix operand order repaired and checked by reading, not compiled, not run on hardware
//
#include <hip/hip_runtime.h>
#include <math.h>

// ---------------- problem constants (match reference) ----------------
#define GN    100000      // nodes
#define GE    1600000     // edges per hop
#define GK    3           // hops
#define GIN   128
#define GOUT  128
#define SLOPE 0.01f

typedef float     v2f  __attribute__((ext_vector_type(2)));
typedef float     v8f  __attribute__((ext_vector_type(8)));
typedef _Float16  v16h __attribute__((ext_vector_type(16)));

#if defined(__has_builtin)
#if __has_builtin(__builtin_amdgcn_wmma_f32_16x16x4_f32)
#define HAVE_WMMA_F32 1
#endif
#endif

// ---------------------------------------------------------------------
// zero-fill (graph-capture safe alternative to hipMemsetAsync)
// ---------------------------------------------------------------------
__global__ __launch_bounds__(256) void zero_f4(float4* __restrict__ p, int n4) {
    int i = blockIdx.x * blockDim.x + threadIdx.x;
    if (i < n4) p[i] = make_float4(0.f, 0.f, 0.f, 0.f);
}

// ---------------------------------------------------------------------
// Phase 1: edge scatter-aggregate.  agg[dst] += w * x[src]
// One wave32 per edge: 32 lanes x float4 = 128 features.
// src/dst/w are wave-uniform -> scalarize with readfirstlane.  Atomic
// adds emitted as native global_atomic_add_f32 (L2-resident, f32-exact).
// ---------------------------------------------------------------------
__global__ __launch_bounds__(256) void scatter_agg(
    const float* __restrict__ x,
    const int*   __restrict__ src,
    const int*   __restrict__ dst,
    const float* __restrict__ w,
    float*       __restrict__ agg)
{
    const int tid = blockIdx.x * blockDim.x + threadIdx.x;
    const int e   = tid >> 5;          // wave-uniform edge id
    const int c   = tid & 31;          // float4 chunk within feature row
    if (e >= GE) return;

    // stream the edge lists ahead through L2 (global_prefetch_b8)
    __builtin_prefetch(src + e + 16384, 0, 1);
    __builtin_prefetch(dst + e + 16384, 0, 1);

    const int   s  = __builtin_amdgcn_readfirstlane(src[e]);
    const int   d  = __builtin_amdgcn_readfirstlane(dst[e]);
    const float wt = __int_as_float(__builtin_amdgcn_readfirstlane(__float_as_int(w[e])));

    const float4 v = reinterpret_cast<const float4*>(x + (size_t)s * GIN)[c];
    float4 m;
    m.x = v.x * wt; m.y = v.y * wt; m.z = v.z * wt; m.w = v.w * wt;

    float* p = agg + (size_t)d * GIN + c * 4;
    asm volatile("global_atomic_add_f32 %0, %1, off" :: "v"(p + 0), "v"(m.x) : "memory");
    asm volatile("global_atomic_add_f32 %0, %1, off" :: "v"(p + 1), "v"(m.y) : "memory");
    asm volatile("global_atomic_add_f32 %0, %1, off" :: "v"(p + 2), "v"(m.z) : "memory");
    asm volatile("global_atomic_add_f32 %0, %1, off" :: "v"(p + 3), "v"(m.w) : "memory");
}

// ---------------------------------------------------------------------
// Phase 2: H = leaky_relu(agg @ W) with f32 WMMA (16x16x4).
// Tile mapping: mT = blockIdx.x (16 node rows, SHARED by all 8 waves of
// the block), nT = wave-in-block (16 output cols each).  The 16x128 A
// tile (8 KB) is staged once per block into LDS via the gfx1250 async
// LDS-load path (global_load_async_to_lds_b128 / ASYNCcnt), with the
// row pitch padded to 132 floats: 528 B keeps the 16 B alignment the
// b128 async writes need and spreads lane reads across banks (4r mod 64).
// Each lane then pulls its v2f A fragment with one conflict-free
// ds_load_b64.  B (64 KB of W, shared device-wide) stays L2-resident.
// Layouts per ISA 7.12.2:
//   A 16x4 f32 : lane(0-15)=row M, vgpr0 = K (+2 for upper half), vgpr1 = K+1
//   B 4x16 f32 : lane%16 = col N, same K split across lane halves
//   C/D 16x16  : vgpr i -> M = i (lanes 0-15) / i+8 (lanes 16-31), N = lane%16
// ---------------------------------------------------------------------
#define APITCH 132   // padded row pitch (floats): 528 B, 16B-aligned

__global__ __launch_bounds__(256) void gemm_lrelu(
    const float* __restrict__ A,    // agg [GN, GIN]
    const float* __restrict__ W,    // [GIN, GOUT]
    float*       __restrict__ H)    // [GN, GOUT] (this hop)
{
    __shared__ float Asf[16 * APITCH];          // 8448 B

    const int lane = threadIdx.x & 31;
    const int half = lane >> 4;
    const int r    = lane & 15;
    const int mT   = blockIdx.x;                // 0..6249
    const int nT   = threadIdx.x >> 5;          // 0..7 (wave in block)

    // ---- stage the 16x128 A tile into LDS (async, ASYNCcnt-tracked) ----
#if defined(__HIP_DEVICE_COMPILE__)
    #pragma unroll
    for (int j = 0; j < 2; ++j) {
        const int idx4 = threadIdx.x + 256 * j;           // 0..511 b128 chunks
        const int rr   = idx4 >> 5;                       // row 0..15
        const int cc   = (idx4 & 31) * 4;                 // col 0,4,...,124
        const float* g = A + (size_t)(mT * 16 + rr) * GIN + cc;
        const unsigned dstoff = (unsigned)(uintptr_t)
            (__attribute__((address_space(3))) float*)(&Asf[rr * APITCH + cc]);
        asm volatile("global_load_async_to_lds_b128 %0, %1, off"
                     :: "v"(dstoff), "v"(g) : "memory");
    }
    asm volatile("s_wait_asynccnt 0" ::: "memory");
#else
    // host-side sema placeholder (never executed)
    for (int j = 0; j < 2; ++j) {
        const int idx4 = threadIdx.x + 256 * j;
        const int rr = idx4 >> 5, cc = (idx4 & 31) * 4;
        for (int q = 0; q < 4; ++q)
            Asf[rr * APITCH + cc + q] = A[(size_t)(mT * 16 + rr) * GIN + cc + q];
    }
#endif
    __syncthreads();

    const float* arow = &Asf[r * APITCH];                  // this lane's A row (LDS)
    const float* bcol = W + nT * 16 + r;                   // this lane's B column

    v8f acc = {};

#ifdef HAVE_WMMA_F32
    #pragma unroll
    for (int k0 = 0; k0 < GIN; k0 += 4) {
        v2f a, b;
        a = *reinterpret_cast<const v2f*>(arow + k0 + 2 * half);   // ds_load_b64
        b.x = bcol[(size_t)(k0 + 2 * half)     * GOUT];
        b.y = bcol[(size_t)(k0 + 2 * half + 1) * GOUT];
        acc = __builtin_amdgcn_wmma_f32_16x16x4_f32(
            /*neg_a=*/false, a, /*neg_b=*/false, b,
            /*c_mod=*/(short)0, acc, /*reuse_a=*/false, /*reuse_b=*/false);
    }
#else
    // Fallback: f16 WMMA 16x16x32 (codegen-confirmed builtin); also the
    // path taken by the host-side semantic pass.
    #pragma unroll
    for (int k0 = 0; k0 < GIN; k0 += 32) {
        v16h a, b;
        #pragma unroll
        for (int e2 = 0; e2 < 16; ++e2) {
            const int vv = e2 >> 1, pp = e2 & 1;
            const int ka = k0 + (vv >> 2) * 16 + (vv & 3) * 2 + pp + 8 * half;
            const int kb = k0 + (vv >> 2) * 16 + (vv & 3) * 2 + pp + 8 * half;
            a[e2] = (_Float16)arow[ka];
            b[e2] = (_Float16)bcol[(size_t)kb * GOUT];
        }
        acc = __builtin_amdgcn_wmma_f32_16x16x32_f16(
            false, a, false, b, (short)0, acc, false, false);
    }
#endif

    #pragma unroll
    for (int i = 0; i < 8; ++i) {
        float vv = acc[i];
        vv = vv > 0.f ? vv : vv * SLOPE;                    // leaky_relu
        H[(size_t)(mT * 16 + i + 8 * half) * GOUT + nT * 16 + r] = vv;
    }
}

// ---------------------------------------------------------------------
// Phase 3: per-node hop attention.  One wave per node; 32 lanes x float4.
// score_k = dot(h_k, att_w) + att_b ; softmax over K=3 ; out = sum a_k h_k
// ---------------------------------------------------------------------
__global__ __launch_bounds__(256) void hop_attention(
    const float* __restrict__ H,      // [GK, GN, GOUT]
    const float* __restrict__ attw,   // [GOUT]
    const float* __restrict__ attb,   // [1]
    float*       __restrict__ out)    // [GN, GOUT]
{
    const int wave = (blockIdx.x * blockDim.x + threadIdx.x) >> 5;  // node
    const int lane = threadIdx.x & 31;
    if (wave >= GN) return;
    const int f = lane * 4;

    const float4 aw = *reinterpret_cast<const float4*>(attw + f);
    float4 hv[GK];
    float  sc[GK];

    #pragma unroll
    for (int k = 0; k < GK; ++k) {
        hv[k] = *reinterpret_cast<const float4*>(
            H + ((size_t)k * GN + wave) * GOUT + f);
        float s = hv[k].x * aw.x + hv[k].y * aw.y + hv[k].z * aw.z + hv[k].w * aw.w;
        s += __shfl_xor(s, 16, 32);       // wave32 butterfly reduction
        s += __shfl_xor(s,  8, 32);
        s += __shfl_xor(s,  4, 32);
        s += __shfl_xor(s,  2, 32);
        s += __shfl_xor(s,  1, 32);
        sc[k] = s + attb[0];
    }

    const float mx = fmaxf(sc[0], fmaxf(sc[1], sc[2]));
    const float e0 = __expf(sc[0] - mx);
    const float e1 = __expf(sc[1] - mx);
    const float e2 = __expf(sc[2] - mx);
    const float inv = 1.0f / (e0 + e1 + e2);
    const float a0 = e0 * inv, a1 = e1 * inv, a2 = e2 * inv;

    float4 o;
    o.x = a0 * hv[0].x + a1 * hv[1].x + a2 * hv[2].x;
    o.y = a0 * hv[0].y + a1 * hv[1].y + a2 * hv[2].y;
    o.z = a0 * hv[0].z + a1 * hv[1].z + a2 * hv[2].z;
    o.w = a0 * hv[0].w + a1 * hv[1].w + a2 * hv[2].w;
    *reinterpret_cast<float4*>(out + (size_t)wave * GOUT + f) = o;
}

// ---------------------------------------------------------------------
extern "C" void kernel_launch(void* const* d_in, const int* in_sizes, int n_in,
                              void* d_out, int out_size, void* d_ws, size_t ws_size,
                              hipStream_t stream)
{
    const float* x    = (const float*)d_in[0];   // [GN, GIN]
    const int*   esrc = (const int*)  d_in[1];   // [GK, GE]
    const int*   edst = (const int*)  d_in[2];   // [GK, GE]
    const float* ew   = (const float*)d_in[3];   // [GK, GE]
    const float* W    = (const float*)d_in[4];   // [GIN, GOUT]
    const float* attw = (const float*)d_in[5];   // [GOUT]
    const float* attb = (const float*)d_in[6];   // [1]
    float*       out  = (float*)d_out;

    // workspace: agg (51.2 MB, reused per hop) then H[K][N][OUT] (153.6 MB)
    float* agg = (float*)d_ws;
    float* H   = agg + (size_t)GN * GIN;

    const int zeroN4     = GN * GIN / 4;                  // 3.2M float4
    const int zeroBlocks = (zeroN4 + 255) / 256;          // 12500
    const int scatBlocks = (GE * 32) / 256;               // 200000 (exact)
    const int gemmBlocks = GN / 16;                       // 6250 (1 mT tile/blk, 8 nT waves)
    const int attnBlocks = (GN + 7) / 8;                  // 12500

    for (int k = 0; k < GK; ++k) {
        zero_f4<<<zeroBlocks, 256, 0, stream>>>((float4*)agg, zeroN4);
        scatter_agg<<<scatBlocks, 256, 0, stream>>>(
            x, esrc + (size_t)k * GE, edst + (size_t)k * GE, ew + (size_t)k * GE, agg);
        gemm_lrelu<<<gemmBlocks, 256, 0, stream>>>(
            agg, W, H + (size_t)k * GN * GOUT);
    }
    hop_attention<<<attnBlocks, 256, 0, stream>>>(H, attw, attb, out);
}
